// MultiHeadAttentionBlock_22548578304648
// MI455X (gfx1250) — compile-verified, matrix operand order checked by reading, not run on hardware
//
#include <hip/hip_runtime.h>

// ---------------------------------------------------------------------------
// MultiHeadAttentionBlock for MI455X (gfx1250, wave32, WMMA bf16)
//   B=2, S=2048, D_MODEL=1024, HEAD=16, D_K=64
// Pipeline:
//   1) Q = q@Wq^T, K = k@Wk^T   -> bf16 [B,H,S,64]
//      V = v@Wv^T (transposed)  -> bf16 [B,H,64,S]   (WMMA B-operand layout)
//   2) flash-attention fused kernel (online softmax, mask honored via L2,
//      K/V tiles staged with GLOBAL_LOAD_ASYNC_TO_LDS when available)
//      -> bf16 [B,S,1024]
//   3) out = attn@Wo^T          -> f32 d_out
// ---------------------------------------------------------------------------

#define DMODEL 1024
#define SEQ    2048
#define NBATCH 2
#define HEADS  16
#define DKV    64
#define MTOT   (NBATCH * SEQ)   // 4096 rows in every GEMM

typedef __bf16 bf16_t;
typedef __attribute__((ext_vector_type(16))) __bf16 v16bf;
typedef __attribute__((ext_vector_type(8)))  float  v8f;
typedef __attribute__((ext_vector_type(4)))  unsigned int uint4v;
typedef __attribute__((ext_vector_type(4)))  int    int4v;
typedef __attribute__((ext_vector_type(4)))  float  float4v;

#if defined(__has_builtin)
#if __has_builtin(__builtin_amdgcn_global_load_async_to_lds_b128) && \
    __has_builtin(__builtin_amdgcn_s_wait_asynccnt)
#define HAVE_ASYNC_LDS 1
#endif
#endif
#ifndef HAVE_ASYNC_LDS
#define HAVE_ASYNC_LDS 0
#endif

#if HAVE_ASYNC_LDS
// builtin signature (from hipcc diagnostic): first arg is non-const
// int4-vector pointer in the device (AS1) address space; LDS side is AS3.
typedef __attribute__((address_space(1))) int4v* gas_p;
typedef __attribute__((address_space(3))) int4v* las_p;
__device__ __forceinline__ void async_cp16(const void* g, void* l) {
    __builtin_amdgcn_global_load_async_to_lds_b128((gas_p)g, (las_p)l, 0, 0);
}
#endif

// Load one WMMA 16x16x32 bf16 operand fragment.
// Per ISA layout: lane holds row (lane&15); lanes<16 take K-octet at k0,
// lanes>=16 at k0+8 (caller folds the +8 into k0); second b128 covers K+16.
__device__ __forceinline__ v16bf ldfrag(const bf16_t* rowp, int k0) {
    union { uint4v u[2]; v16bf v; } f;
    f.u[0] = *(const uint4v*)(rowp + k0);
    f.u[1] = *(const uint4v*)(rowp + k0 + 16);
    return f.v;
}

__device__ __forceinline__ v8f wmma_bf16(v16bf a, v16bf b, v8f c) {
    return __builtin_amdgcn_wmma_f32_16x16x32_bf16(
        /*neg_a=*/false, a, /*neg_b=*/false, b,
        /*c_mod=*/(short)0, c, /*reuse_a=*/false, /*reuse_b=*/false);
}

// 16 contiguous elements -> bf16 regs (f32 source: 4x b128 loads + cvt)
__device__ __forceinline__ void ld16(const float* src, bf16_t* dst) {
    const float4v* s4 = (const float4v*)src;
#pragma unroll
    for (int i = 0; i < 4; ++i) {
        float4v t = s4[i];
        dst[i * 4 + 0] = (bf16_t)t.x;
        dst[i * 4 + 1] = (bf16_t)t.y;
        dst[i * 4 + 2] = (bf16_t)t.z;
        dst[i * 4 + 3] = (bf16_t)t.w;
    }
}
// 16 contiguous bf16 -> regs (2x b128 loads)
__device__ __forceinline__ void ld16(const bf16_t* src, bf16_t* dst) {
    union { uint4v u; bf16_t h[8]; } a, b;
    a.u = ((const uint4v*)src)[0];
    b.u = ((const uint4v*)src)[1];
#pragma unroll
    for (int i = 0; i < 8; ++i) { dst[i] = a.h[i]; dst[8 + i] = b.h[i]; }
}
// pack 8 bf16 regs -> one b128 LDS store
__device__ __forceinline__ void st8(bf16_t* dst, const bf16_t* src) {
    union { bf16_t h[8]; uint4v u; } p;
#pragma unroll
    for (int i = 0; i < 8; ++i) p.h[i] = src[i];
    *(uint4v*)dst = p.u;
}

// ---------------------------------------------------------------------------
// GEMM: C[M,N] = A[M,K] @ W[N,K]^T     (M=4096, N=K=1024)
// Workgroup tile 128x128, 8 waves of 32x64, K-step 32, bf16 WMMA, f32 acc.
// MODE 0: store bf16 [B,H,S,64]   (Q,K)
// MODE 1: store bf16 [B,H,64,S]   (V transposed)
// MODE 2: store f32  [M,1024]     (final projection -> d_out)
// ---------------------------------------------------------------------------
template <typename TA, int MODE>
__global__ __launch_bounds__(256)
void gemm_xWt(const TA* __restrict__ A, const float* __restrict__ W,
              void* __restrict__ Out) {
    // 32-wide K tiles, row stride padded to 40 bf16 (80 B): 16B-aligned,
    // conflict-free for 16-lane row-parallel b128 reads (20*r mod 64 distinct).
    __shared__ __align__(16) bf16_t As[128][40];
    __shared__ __align__(16) bf16_t Ws[128][40];

    const int tid  = threadIdx.x;
    const int lane = tid & 31;
    const int w    = tid >> 5;
    const int mBase = blockIdx.y * 128;
    const int nBase = blockIdx.x * 128;

    // staging: each thread fills 16 elements of one row (128 rows x 2 halves)
    const int ldRow = tid >> 1;
    const int ldCol = (tid & 1) * 16;

    const int mRow   = lane & 15;
    const int hsel   = (lane < 16) ? 0 : 8;   // K-octet select per half-wave
    const int rowAdd = (lane >= 16) ? 8 : 0;  // C-layout row offset

    const int wm = (w & 3) * 32;   // wave M offset (4 tiles of 32)
    const int wn = (w >> 2) * 64;  // wave N offset (2 tiles of 64)

    v8f c[2][4];
#pragma unroll
    for (int i = 0; i < 2; ++i)
#pragma unroll
        for (int j = 0; j < 4; ++j) c[i][j] = {};

    const size_t gmOff = (size_t)(mBase + ldRow) * DMODEL;
    const size_t gnOff = (size_t)(nBase + ldRow) * DMODEL;

    bf16_t aStage[16], wStage[16];
    ld16(A + gmOff + ldCol, aStage);
    ld16(W + gnOff + ldCol, wStage);

    for (int kt = 0; kt < DMODEL / 32; ++kt) {
        st8(&As[ldRow][ldCol],     aStage);
        st8(&As[ldRow][ldCol + 8], aStage + 8);
        st8(&Ws[ldRow][ldCol],     wStage);
        st8(&Ws[ldRow][ldCol + 8], wStage + 8);
        __syncthreads();

        if (kt + 1 < DMODEL / 32) {  // prefetch next K tile into registers
            ld16(A + gmOff + (kt + 1) * 32 + ldCol, aStage);
            ld16(W + gnOff + (kt + 1) * 32 + ldCol, wStage);
        }

        v16bf af[2], bfg[4];
#pragma unroll
        for (int i = 0; i < 2; ++i)
            af[i] = ldfrag(&As[wm + i * 16 + mRow][0], hsel);
#pragma unroll
        for (int j = 0; j < 4; ++j)
            bfg[j] = ldfrag(&Ws[wn + j * 16 + mRow][0], hsel);

#pragma unroll
        for (int i = 0; i < 2; ++i)
#pragma unroll
            for (int j = 0; j < 4; ++j)
                c[i][j] = wmma_bf16(af[i], bfg[j], c[i][j]);

        __syncthreads();
    }

    // epilogue
#pragma unroll
    for (int i = 0; i < 2; ++i) {
#pragma unroll
        for (int j = 0; j < 4; ++j) {
#pragma unroll
            for (int r = 0; r < 8; ++r) {
                const int row = mBase + wm + i * 16 + r + rowAdd;
                const int col = nBase + wn + j * 16 + mRow;
                const float val = c[i][j][r];
                if (MODE == 2) {
                    ((float*)Out)[(size_t)row * DMODEL + col] = val;
                } else {
                    const int b = row / SEQ, s = row % SEQ;
                    const int h = col / DKV, d = col % DKV;
                    if (MODE == 0)
                        ((bf16_t*)Out)[(((size_t)(b * HEADS + h)) * SEQ + s) * DKV + d] =
                            (bf16_t)val;
                    else
                        ((bf16_t*)Out)[(((size_t)(b * HEADS + h)) * DKV + d) * SEQ + s] =
                            (bf16_t)val;
                }
            }
        }
    }
}

// ---------------------------------------------------------------------------
// Fused flash attention: one workgroup = one (b,h) x 128-query block.
// Each wave owns 16 queries; 64-key blocks staged in LDS; online softmax.
// ---------------------------------------------------------------------------
__global__ __launch_bounds__(256)
void attn_flash(const bf16_t* __restrict__ Qb, const bf16_t* __restrict__ Kb,
                const bf16_t* __restrict__ Vtb, const int* __restrict__ mask,
                bf16_t* __restrict__ Ob) {
    // 64-wide tiles, row stride padded to 72 bf16 (144 B): 16B-aligned,
    // conflict-free (36*r mod 64 distinct over 16 rows).
    __shared__ __align__(16) bf16_t Ks[64][72];   // [key][d]
    __shared__ __align__(16) bf16_t Vs[64][72];   // [d][key]  (V stored transposed)
    __shared__ __align__(16) bf16_t Ps[8][16][72];// per-wave P patch [qrow][key]

    const int tid  = threadIdx.x;
    const int lane = tid & 31;
    const int w    = tid >> 5;
    const int bh   = blockIdx.y;                 // b*HEADS + h
    const int qw   = blockIdx.x * 128 + w * 16;  // wave's first query row

    const bf16_t* Qp = Qb  + (size_t)bh * SEQ * DKV;
    const bf16_t* Kp = Kb  + (size_t)bh * SEQ * DKV;
    const bf16_t* Vp = Vtb + (size_t)bh * DKV * SEQ;

    const int mRow   = lane & 15;
    const int hsel   = (lane < 16) ? 0 : 8;
    const int rowAdd = (lane >= 16) ? 8 : 0;
    const float scale = 0.125f;   // 1/sqrt(64)

    // Q fragments (A-operand), loaded once straight from global
    v16bf qf[2];
    {
        const bf16_t* qrow = Qp + (size_t)(qw + mRow) * DKV;
        qf[0] = ldfrag(qrow, 0 + hsel);
        qf[1] = ldfrag(qrow, 32 + hsel);
    }

    v8f o[4];
#pragma unroll
    for (int j = 0; j < 4; ++j) o[j] = {};
    float mstate[8], lstate[8];
#pragma unroll
    for (int r = 0; r < 8; ++r) { mstate[r] = -3.0e38f; lstate[r] = 0.0f; }

    // staging: 64 rows x 64 bf16, 4 threads per row, 16 elems (2x b128) each
    const int sRow = tid >> 2;
    const int sCol = (tid & 3) * 16;

    for (int kb = 0; kb < SEQ / 64; ++kb) {
        const int keyBase = kb * 64;
        {
            const bf16_t* kg = Kp + (size_t)(keyBase + sRow) * DKV + sCol;
            const bf16_t* vg = Vp + (size_t)sRow * SEQ + keyBase + sCol;
#if HAVE_ASYNC_LDS
            // CDNA5 async copy: global -> LDS without VGPR round-trip,
            // tracked on ASYNCcnt (s_wait_asynccnt before the barrier).
            async_cp16(kg,     &Ks[sRow][sCol]);
            async_cp16(kg + 8, &Ks[sRow][sCol + 8]);
            async_cp16(vg,     &Vs[sRow][sCol]);
            async_cp16(vg + 8, &Vs[sRow][sCol + 8]);
            __builtin_amdgcn_s_wait_asynccnt(0);
#else
            const uint4v* ksrc = (const uint4v*)kg;
            uint4v k0 = ksrc[0], k1 = ksrc[1];
            *(uint4v*)&Ks[sRow][sCol]     = k0;
            *(uint4v*)&Ks[sRow][sCol + 8] = k1;
            const uint4v* vsrc = (const uint4v*)vg;
            uint4v v0 = vsrc[0], v1 = vsrc[1];
            *(uint4v*)&Vs[sRow][sCol]     = v0;
            *(uint4v*)&Vs[sRow][sCol + 8] = v1;
#endif
        }
        __syncthreads();

        // scores: S[16 x 64] = Q[16x64] @ K_block^T, as 4 N-tiles x 2 K-steps
        v8f s[4];
#pragma unroll
        for (int j = 0; j < 4; ++j) {
            v16bf kf0 = ldfrag(&Ks[j * 16 + mRow][0], hsel);
            v16bf kf1 = ldfrag(&Ks[j * 16 + mRow][0], 32 + hsel);
            v8f acc = {};
            acc = wmma_bf16(qf[0], kf0, acc);
            acc = wmma_bf16(qf[1], kf1, acc);
            s[j] = acc;
        }

        // scale + mask + block row-max (rows live in 16-lane half-waves)
        float blkmax[8];
#pragma unroll
        for (int r = 0; r < 8; ++r) {
            const int qrow = qw + r + rowAdd;
            float mx = -3.0e38f;
#pragma unroll
            for (int j = 0; j < 4; ++j) {
                const int kcol = keyBase + j * 16 + mRow;
                const int mv = mask[(size_t)qrow * SEQ + kcol];
                float sv = s[j][r] * scale;
                sv = (mv == 0) ? -1.0e9f : sv;
                s[j][r] = sv;
                mx = fmaxf(mx, sv);
            }
#pragma unroll
            for (int d = 1; d < 16; d <<= 1)
                mx = fmaxf(mx, __shfl_xor(mx, d, 32));
            blkmax[r] = mx;
        }

        // online softmax update
#pragma unroll
        for (int r = 0; r < 8; ++r) {
            const float mnew  = fmaxf(mstate[r], blkmax[r]);
            const float alpha = __expf(mstate[r] - mnew);
            mstate[r] = mnew;
            float rsum = 0.0f;
#pragma unroll
            for (int j = 0; j < 4; ++j) {
                const float p = __expf(s[j][r] - mnew);
                s[j][r] = p;
                rsum += p;
            }
#pragma unroll
            for (int d = 1; d < 16; d <<= 1)
                rsum += __shfl_xor(rsum, d, 32);
            lstate[r] = lstate[r] * alpha + rsum;
#pragma unroll
            for (int j = 0; j < 4; ++j) o[j][r] *= alpha;
        }

        // C-layout -> A-layout via per-wave LDS patch
#pragma unroll
        for (int j = 0; j < 4; ++j)
#pragma unroll
            for (int r = 0; r < 8; ++r)
                Ps[w][r + rowAdd][j * 16 + mRow] = (bf16_t)s[j][r];

        // O += P[16x64] @ V_block[64x64]
        v16bf pf0 = ldfrag(&Ps[w][mRow][0], hsel);
        v16bf pf1 = ldfrag(&Ps[w][mRow][0], 32 + hsel);
#pragma unroll
        for (int j = 0; j < 4; ++j) {
            v16bf vf0 = ldfrag(&Vs[j * 16 + mRow][0], hsel);
            v16bf vf1 = ldfrag(&Vs[j * 16 + mRow][0], 32 + hsel);
            o[j] = wmma_bf16(pf0, vf0, o[j]);
            o[j] = wmma_bf16(pf1, vf1, o[j]);
        }
        __syncthreads();   // before next block overwrites Ks/Vs
    }

    // epilogue: O /= L, store bf16 [B, S, H*DK]
    const int b = bh / HEADS, h = bh % HEADS;
#pragma unroll
    for (int j = 0; j < 4; ++j) {
#pragma unroll
        for (int r = 0; r < 8; ++r) {
            const int qrow = qw + r + rowAdd;
            const int dcol = j * 16 + mRow;
            const float val = o[j][r] / lstate[r];
            Ob[((size_t)(b * SEQ + qrow)) * DMODEL + h * DKV + dcol] = (bf16_t)val;
        }
    }
}

// ---------------------------------------------------------------------------
extern "C" void kernel_launch(void* const* d_in, const int* in_sizes, int n_in,
                              void* d_out, int out_size, void* d_ws, size_t ws_size,
                              hipStream_t stream) {
    const float* q    = (const float*)d_in[0];
    const float* k    = (const float*)d_in[1];
    const float* v    = (const float*)d_in[2];
    const int*   mask = (const int*)  d_in[3];
    const float* wq   = (const float*)d_in[4];
    const float* wk   = (const float*)d_in[5];
    const float* wv   = (const float*)d_in[6];
    const float* wo   = (const float*)d_in[7];
    float* out = (float*)d_out;

    // workspace: 4 x 8 MiB bf16 buffers
    char* ws = (char*)d_ws;
    const size_t BUF = (size_t)MTOT * DMODEL * sizeof(bf16_t);  // 8 MiB
    bf16_t* Qb = (bf16_t*)(ws + 0 * BUF);
    bf16_t* Kb = (bf16_t*)(ws + 1 * BUF);
    bf16_t* Vb = (bf16_t*)(ws + 2 * BUF);
    bf16_t* Ab = (bf16_t*)(ws + 3 * BUF);

    dim3 gg(DMODEL / 128, MTOT / 128);  // (8, 32)
    gemm_xWt<float, 0><<<gg, 256, 0, stream>>>(q, wq, Qb);
    gemm_xWt<float, 0><<<gg, 256, 0, stream>>>(k, wk, Kb);
    gemm_xWt<float, 1><<<gg, 256, 0, stream>>>(v, wv, Vb);

    attn_flash<<<dim3(SEQ / 128, NBATCH * HEADS), 256, 0, stream>>>(
        Qb, Kb, Vb, mask, Ab);

    gemm_xWt<bf16_t, 2><<<gg, 256, 0, stream>>>(Ab, wo, out);
}
